// CombinedFocalLoss_87634512707758
// MI455X (gfx1250) — matrix operand order repaired and checked by reading, not compiled, not run on hardware
//
#include <hip/hip_runtime.h>
#include <hip/hip_bf16.h>
#include <math.h>

typedef __attribute__((ext_vector_type(16))) _Float16 v16h;
typedef __attribute__((ext_vector_type(8)))  float    v8f;
typedef __attribute__((ext_vector_type(4)))  float    v4f;

#define BATCH 16
#define HW    36864      // 192*192
#define DIM   64
#define EPSC  0.0001f

// workspace float layout
#define WS_POS    0
#define WS_NEG    1
#define WS_NPOS   2
#define WS_OFFSQ  3
#define WS_NCOEF  4
#define WS_CSTSQ  5
#define WS_NORMSQ 8      // 16 floats
#define WS_GRAM   32     // 256 floats
#define WS_FEAT   288    // 1024 floats
#define WS_KEYS_FOFF 1312  // ull[16] at byte offset 1312*4 (8B aligned)

__device__ __forceinline__ float block_reduce_sum(float v, float* sm) {
    int tid = threadIdx.x;
    sm[tid] = v; __syncthreads();
    for (int s = blockDim.x >> 1; s > 0; s >>= 1) {
        if (tid < s) sm[tid] += sm[tid + s];
        __syncthreads();
    }
    float r = sm[0];
    __syncthreads();
    return r;
}

// ---------------- init ----------------
__global__ void init_ws_kernel(float* f, unsigned long long* keys) {
    int t = threadIdx.x;
    if (t < 288) f[t] = 0.0f;           // scalars + normsq + gram
    if (t < 16)  keys[t] = 0ull;
}

// ---------------- focal loss + per-row norm^2 ----------------
// grid (36, 16), block 256; 4 elements per thread (16B vector loads)
__global__ void hm_kernel(const float* __restrict__ x, const float* __restrict__ t,
                          float* ws) {
    __shared__ float sm[256];
    int b = blockIdx.y;
    size_t i4 = (size_t)b * (HW / 4) + blockIdx.x * 256 + threadIdx.x;
    v4f xv = *((const v4f*)x + i4);                      // reused by gram -> keep cached
    v4f gt = __builtin_nontemporal_load((const v4f*)t + i4);  // read-once stream

    float posl = 0.0f, negl = 0.0f, npos = 0.0f, nsq = 0.0f;
#pragma unroll
    for (int e = 0; e < 4; ++e) {
        float xe = xv[e], ge = gt[e];
        float p  = 1.0f / (1.0f + __expf(-xe));
        p = fminf(fmaxf(p, EPSC), 1.0f - EPSC);
        float pos  = (ge == 1.0f) ? 1.0f : 0.0f;
        float neg  = (ge <  1.0f) ? 1.0f : 0.0f;
        float om   = 1.0f - ge;
        float negw = om * om * om * om;
        float omp  = 1.0f - p;
        posl += 0.92f * __logf(p) * omp * omp * pos;     // 1 - FL_EPS + FL_EPS*NOISE
        negl += __logf(1.0f - p) * p * p * neg * negw;
        npos += pos;
        nsq  += xe * xe;
    }
    float r;
    r = block_reduce_sum(posl, sm); if (threadIdx.x == 0) atomicAdd(&ws[WS_POS], r);
    r = block_reduce_sum(negl, sm); if (threadIdx.x == 0) atomicAdd(&ws[WS_NEG], r);
    r = block_reduce_sum(npos, sm); if (threadIdx.x == 0) atomicAdd(&ws[WS_NPOS], r);
    r = block_reduce_sum(nsq, sm);  if (threadIdx.x == 0) atomicAdd(&ws[WS_NORMSQ + b], r);
}

// ---------------- Gram matrix G = U U^T via v_wmma_f32_16x16x32_f16 ----------------
// 36 blocks x 32 threads (1 wave); block covers K range of 1024; two independent
// accumulators (K interleaved by 32) to break the WMMA D->C dependency chain.
__global__ void gram_wmma_kernel(const float* __restrict__ hm, float* gram) {
    const int lane = threadIdx.x;      // 0..31
    const int m    = lane & 15;
    const int half = lane >> 4;
    const float* row = hm + (size_t)m * HW;   // A: M = lane&15 ; B: N = lane&15 (B = U^T)

    v8f c0 = {}, c1 = {};
    const int kbase = blockIdx.x * 1024;
    for (int k0 = kbase; k0 < kbase + 1024; k0 += 64) {
        v16h a0, b0, a1, b1;
#pragma unroll
        for (int j = 0; j < 8; ++j) {
            // A 16x32 f16 layout: VGPR j<4 -> K = 2j + 8*half ; j>=4 -> K = 16 + 2(j-4) + 8*half
            int kA = (j < 4) ? (2 * j + 8 * half) : (16 + 2 * (j - 4) + 8 * half);
            // B 32x16 f16 layout: same K mapping as A
            int kB = kA;
            float2 fa0 = *(const float2*)(row + k0 + kA);
            float2 fb0 = *(const float2*)(row + k0 + kB);
            float2 fa1 = *(const float2*)(row + k0 + 32 + kA);
            float2 fb1 = *(const float2*)(row + k0 + 32 + kB);
            a0[2 * j] = (_Float16)fa0.x; a0[2 * j + 1] = (_Float16)fa0.y;
            b0[2 * j] = (_Float16)fb0.x; b0[2 * j + 1] = (_Float16)fb0.y;
            a1[2 * j] = (_Float16)fa1.x; a1[2 * j + 1] = (_Float16)fa1.y;
            b1[2 * j] = (_Float16)fb1.x; b1[2 * j + 1] = (_Float16)fb1.y;
        }
        c0 = __builtin_amdgcn_wmma_f32_16x16x32_f16(false, a0, false, b0,
                                                    (short)0, c0, false, false);
        c1 = __builtin_amdgcn_wmma_f32_16x16x32_f16(false, a1, false, b1,
                                                    (short)0, c1, false, false);
    }
    v8f c = c0 + c1;
#pragma unroll
    for (int r = 0; r < 8; ++r) {
        int M = r + 8 * half;                 // C/D layout: M = vgpr + 8*half, N = lane&15
        atomicAdd(&gram[M * 16 + m], c[r]);
    }
}

// ---------------- masked offset MSE ----------------
// grid 1152, block 256; 4 elements/thread over 16*2*192*192
__global__ void off_kernel(const float* __restrict__ op, const float* __restrict__ og,
                           float* ws) {
    __shared__ float sm[256];
    size_t i4 = (size_t)blockIdx.x * 256 + threadIdx.x;
    v4f pv = __builtin_nontemporal_load((const v4f*)op + i4);
    v4f gv = __builtin_nontemporal_load((const v4f*)og + i4);
    float sq = 0.0f, cnt = 0.0f;
#pragma unroll
    for (int e = 0; e < 4; ++e) {
        float c = (gv[e] > 0.0f) ? 1.0f : 0.0f;
        float d = (pv[e] - gv[e]) * c;
        sq += d * d; cnt += c;
    }
    float r;
    r = block_reduce_sum(sq, sm);  if (threadIdx.x == 0) atomicAdd(&ws[WS_OFFSQ], r);
    r = block_reduce_sum(cnt, sm); if (threadIdx.x == 0) atomicAdd(&ws[WS_NCOEF], r);
}

// ---------------- per-batch argmax of cstency_gts ----------------
// key = float_bits<<32 | ~index  (valid: values non-negative; first-index tie-break)
__global__ void argmax_kernel(const float* __restrict__ g, unsigned long long* keys) {
    __shared__ unsigned long long sm[256];
    int b = blockIdx.y;
    int i0 = (blockIdx.x * 256 + threadIdx.x) * 4;
    v4f v = *((const v4f*)(g + (size_t)b * HW + i0));    // gts reused by cst -> keep cached
    unsigned long long key = 0ull;
#pragma unroll
    for (int e = 0; e < 4; ++e) {
        unsigned long long k =
            ((unsigned long long)__float_as_uint(v[e]) << 32) |
            (unsigned long long)(0xFFFFFFFFu - (unsigned)(i0 + e));
        key = (k > key) ? k : key;
    }
    int tid = threadIdx.x;
    sm[tid] = key; __syncthreads();
    for (int s = 128; s > 0; s >>= 1) {
        if (tid < s) sm[tid] = (sm[tid] > sm[tid + s]) ? sm[tid] : sm[tid + s];
        __syncthreads();
    }
    if (tid == 0) atomicMax(&keys[b], sm[0]);
}

// ---------------- gather feat[b][d] = preds[b, d, idx[b]] ----------------
__global__ void feat_kernel(const float* __restrict__ preds,
                            const unsigned long long* __restrict__ keys,
                            float* feat) {
    int t = blockIdx.x * 256 + threadIdx.x;          // 0..1023
    if (t >= BATCH * DIM) return;
    int b = t >> 6, d = t & 63;
    unsigned idx = 0xFFFFFFFFu - (unsigned)(keys[b] & 0xFFFFFFFFull);
    feat[t] = preds[((size_t)b * DIM + d) * HW + idx];
}

// ---------------- consistency loss: stream 151 MB exactly once (NT) ----------------
// grid (36, 16), block 256; 4 pixels/thread, 16B NT loads
__global__ void cst_kernel(const float* __restrict__ preds,
                           const float* __restrict__ gts,
                           const float* __restrict__ feat, float* ws) {
    __shared__ float fs[DIM];
    __shared__ float sm[256];
    int b = blockIdx.y;
    int tid = threadIdx.x;
    if (tid < DIM) fs[tid] = feat[b * DIM + tid];
    __syncthreads();
    int n4 = blockIdx.x * 256 + tid;                      // float4 index within batch
    const v4f* base = (const v4f*)(preds + (size_t)b * DIM * HW) + n4;
    v4f s = {};
#pragma unroll 8
    for (int d = 0; d < DIM; ++d) {
        v4f p = __builtin_nontemporal_load(base + (size_t)d * (HW / 4));
        s += fs[d] * p;
    }
    v4f gt = *((const v4f*)(gts + (size_t)b * HW) + n4);
    float acc = 0.0f;
#pragma unroll
    for (int e = 0; e < 4; ++e) {
        float enc  = 1.0f / (1.0f + __expf(-s[e] * 0.125f));   // 1/sqrt(64)
        float diff = enc - gt[e];
        acc += diff * diff;
    }
    float r = block_reduce_sum(acc, sm);
    if (tid == 0) atomicAdd(&ws[WS_CSTSQ], r);
}

// ---------------- finalize all six losses ----------------
__global__ void finalize_kernel(const float* __restrict__ cls_p, const float* __restrict__ cls_g,
                                const float* __restrict__ tmp_p, const float* __restrict__ tmp_g,
                                const float* __restrict__ ws, float* out) {
    if (threadIdx.x != 0 || blockIdx.x != 0) return;
    // hm
    float pos_s = ws[WS_POS], neg_s = ws[WS_NEG], npos = ws[WS_NPOS];
    float loss_hm = (npos == 0.0f) ? -neg_s : -(pos_s + neg_s) / fmaxf(npos, 1.0f);
    // cls (BCE mean over 16)
    float s = 0.0f;
    for (int i = 0; i < 16; ++i) {
        float x = cls_p[i], y = cls_g[i];
        s += fmaxf(x, 0.0f) + log1pf(__expf(-fabsf(x))) - x * y;
    }
    float loss_cls = s / 16.0f;
    // dst (normalize Gram by row norms)
    float norms[16];
    for (int b = 0; b < 16; ++b) norms[b] = fmaxf(sqrtf(ws[WS_NORMSQ + b]), 1e-6f);
    float pos_sum = 0.0f, neg_sum = 0.0f;
    for (int i = 0; i < 8; ++i)
        for (int j = 0; j < 16; ++j) {
            float cosv = ws[WS_GRAM + i * 16 + j] / (norms[i] * norms[j]);
            float v = 0.5f * (1.0f - cosv);
            if (j < 8) pos_sum += v; else neg_sum += v;
        }
    float loss_dst = (pos_sum - neg_sum) / 64.0f * 0.1f;
    // off
    float loss_off = 0.5f * (ws[WS_OFFSQ] / 1179648.0f) / (ws[WS_NCOEF] + 1e-6f);
    // cst
    float loss_cst = ws[WS_CSTSQ] / 589824.0f * 0.1f;
    // tmp (BCE mean over 128)
    float s2 = 0.0f;
    for (int i = 0; i < 128; ++i) {
        float x = tmp_p[i], y = tmp_g[i];
        s2 += fmaxf(x, 0.0f) + log1pf(__expf(-fabsf(x))) - x * y;
    }
    float loss_tmp = s2 / 128.0f;

    out[0] = loss_hm; out[1] = loss_cls; out[2] = loss_dst;
    out[3] = loss_off; out[4] = loss_cst; out[5] = loss_tmp;
}

extern "C" void kernel_launch(void* const* d_in, const int* in_sizes, int n_in,
                              void* d_out, int out_size, void* d_ws, size_t ws_size,
                              hipStream_t stream) {
    const float* hm_out  = (const float*)d_in[0];
    const float* hm_tgt  = (const float*)d_in[1];
    const float* cls_p   = (const float*)d_in[2];
    const float* cls_g   = (const float*)d_in[3];
    const float* off_p   = (const float*)d_in[4];
    const float* off_g   = (const float*)d_in[5];
    const float* cst_p   = (const float*)d_in[6];
    const float* cst_g   = (const float*)d_in[7];
    const float* tmp_p   = (const float*)d_in[8];
    const float* tmp_g   = (const float*)d_in[9];

    float* wsf = (float*)d_ws;
    unsigned long long* keys =
        (unsigned long long*)((char*)d_ws + WS_KEYS_FOFF * sizeof(float));
    float* out = (float*)d_out;

    init_ws_kernel<<<1, 512, 0, stream>>>(wsf, keys);
    hm_kernel<<<dim3(36, 16), 256, 0, stream>>>(hm_out, hm_tgt, wsf);
    // run gram while hm_outputs (2.4 MB) is still L2-resident, before the 151 MB stream
    gram_wmma_kernel<<<36, 32, 0, stream>>>(hm_out, wsf + WS_GRAM);
    off_kernel<<<dim3(1152), 256, 0, stream>>>(off_p, off_g, wsf);
    argmax_kernel<<<dim3(36, 16), 256, 0, stream>>>(cst_g, keys);
    feat_kernel<<<4, 256, 0, stream>>>(cst_p, keys, wsf + WS_FEAT);
    cst_kernel<<<dim3(36, 16), 256, 0, stream>>>(cst_p, cst_g, wsf + WS_FEAT, wsf);
    finalize_kernel<<<1, 32, 0, stream>>>(cls_p, cls_g, tmp_p, tmp_g, wsf, out);
}
